// AdapterController_55104430408052
// MI455X (gfx1250) — matrix operand order repaired and checked by reading, not compiled, not run on hardware
//
#include <hip/hip_runtime.h>
#include <hip/hip_bf16.h>

// Problem constants (match reference)
#define MM 4      // routers
#define BB 16     // batch
#define SS 512    // seq
#define CC 1024   // channel
#define DD 256    // down size
#define NN 16     // experts

typedef __attribute__((ext_vector_type(16))) __bf16 v16bf;
typedef __attribute__((ext_vector_type(8)))  float  v8f;
typedef __attribute__((ext_vector_type(4)))  int    v4i;

union Frag {
    v16bf v;
    uint4 q[2];
};

__device__ __forceinline__ unsigned f32_to_bf16_bits(float f) {
    unsigned u = __float_as_uint(f);
    return (u + 0x7FFFu + ((u >> 16) & 1u)) >> 16;
}
__device__ __forceinline__ unsigned pack_bf16x2(float lo, float hi) {
    return f32_to_bf16_bits(lo) | (f32_to_bf16_bits(hi) << 16);
}

// ---------------------------------------------------------------------------
// Kernel 1: z = SiLU(x @ Wd + bd), bf16 output to workspace.
// grid = (M*B, S/64), block = 256 (8 wave32)
// Each wave: 16-row strip x 128 cols of the 64x256 z tile (8 accum tiles).
// ---------------------------------------------------------------------------
__global__ __launch_bounds__(256)
void adapter_down_silu(const float* __restrict__ x,
                       const int* __restrict__ expert_index,
                       const float* __restrict__ down_w,
                       const float* __restrict__ down_b,
                       unsigned short* __restrict__ z_ws)
{
    __shared__ __align__(16) unsigned short xs[64][32];   // A tile, row-major [row][k]
    __shared__ __align__(16) unsigned short wt[256][32];  // B tile, K-contig per column [d][k]

    const int mb = blockIdx.x;          // m*B + b
    const int m  = mb / BB;
    const int b  = mb % BB;
    const int s0 = blockIdx.y * 64;

    const int e = expert_index[m * BB + b];

    // Uniform bases (SGPR) + small int offsets -> SADDR-form global loads
    const float* xb  = x + (b * SS + s0) * CC;
    const float* wd  = down_w + (size_t)(m * NN + e) * (CC * DD);
    const float* bd  = down_b + (m * NN + e) * DD;
    unsigned short* zb = z_ws + (mb * SS + s0) * DD;

    const int t    = threadIdx.x;
    const int lane = t & 31;
    const int w    = t >> 5;
    const int strip    = w & 3;        // which 16-row strip
    const int colbase  = (w >> 2) * 128;

    v8f acc[8];
#pragma unroll
    for (int n = 0; n < 8; ++n) acc[n] = {};

    const int arow = strip * 16 + (lane & 15);
    const int hb   = lane >> 4;

    for (int kc = 0; kc < CC; kc += 32) {
        // Stage x tile [64 x 32]: float2 load -> packed bf16x2 b32 DS store
#pragma unroll
        for (int i = 0; i < 4; ++i) {
            int p   = i * 256 + t;      // float2-pair index, 0..1023
            int row = p >> 4;
            int k2  = p & 15;
            float2 v = *(const float2*)&xb[row * CC + kc + 2 * k2];
            *(unsigned*)&xs[row][2 * k2] = pack_bf16x2(v.x, v.y);
        }
        // Stage Wd chunk [32 x 256] transposed -> wt[d][k].
        // 2x2 patches: two float2 loads (rows k, k+1), two packed b32 stores.
#pragma unroll
        for (int i = 0; i < 8; ++i) {
            int p  = i * 256 + t;       // patch index, 0..2047
            int k2 = p >> 7;            // k-pair 0..15
            int d2 = p & 127;           // d-pair 0..127
            float2 r0 = *(const float2*)&wd[(kc + 2 * k2) * DD + 2 * d2];
            float2 r1 = *(const float2*)&wd[(kc + 2 * k2 + 1) * DD + 2 * d2];
            *(unsigned*)&wt[2 * d2][2 * k2]     = pack_bf16x2(r0.x, r1.x);
            *(unsigned*)&wt[2 * d2 + 1][2 * k2] = pack_bf16x2(r0.y, r1.y);
        }
        // Prefetch next weight chunk into L2 (global_prefetch_b8)
        if (kc + 32 < CC) {
            __builtin_prefetch(&wd[(kc + 32) * DD + t], 0, 1);
        }
        __syncthreads();

        // A fragment: row = arow, K runs [hb*8, hb*8+8) and [hb*8+16, hb*8+24)
        Frag af;
        af.q[0] = *(const uint4*)&xs[arow][hb * 8];
        af.q[1] = *(const uint4*)&xs[arow][hb * 8 + 16];

#pragma unroll
        for (int n = 0; n < 8; ++n) {
            int col = colbase + n * 16 + (lane & 15);
            int kh  = hb * 8;
            Frag bf;
            bf.q[0] = *(const uint4*)&wt[col][kh];
            bf.q[1] = *(const uint4*)&wt[col][kh + 16];
            acc[n] = __builtin_amdgcn_wmma_f32_16x16x32_bf16(
                false, af.v, false, bf.v, (short)0, acc[n], false, false);
        }
        __syncthreads();
    }

    // Epilogue: bias + SiLU, store bf16 z
#pragma unroll
    for (int n = 0; n < 8; ++n) {
        int col = colbase + n * 16 + (lane & 15);
        float bias = bd[col];
#pragma unroll
        for (int v = 0; v < 8; ++v) {
            int row = strip * 16 + v + 8 * hb;
            float val = acc[n][v] + bias;
            val = val / (1.0f + __expf(-val));   // SiLU
            zb[row * DD + col] = (unsigned short)f32_to_bf16_bits(val);
        }
    }
}

// ---------------------------------------------------------------------------
// Kernel 2: u = z @ Wu, f32 output.
// grid = (M*B, S/64, C/256), block = 256 (8 wave32)
// z tile [64 x 256] staged once (async-to-LDS if available); Wu streamed.
// ---------------------------------------------------------------------------
__global__ __launch_bounds__(256)
void adapter_up(const unsigned short* __restrict__ z_ws,
                const int* __restrict__ expert_index,
                const float* __restrict__ up_w,
                float* __restrict__ out)
{
    __shared__ __align__(16) unsigned short zs[64][256];  // A tile, row-major
    __shared__ __align__(16) unsigned short wt[256][32];  // B tile, K-contig per column

    const int mb = blockIdx.x;
    const int m  = mb / BB;
    const int b  = mb % BB;
    const int s0 = blockIdx.y * 64;
    const int c0 = blockIdx.z * 256;

    const int e = expert_index[m * BB + b];

    const unsigned short* zb = z_ws + (mb * SS + s0) * DD;
    const float* wu = up_w + (size_t)(m * NN + e) * (DD * CC);
    float* ub = out + (size_t)(mb * SS + s0) * CC + c0;

    const int t    = threadIdx.x;
    const int lane = t & 31;
    const int w    = t >> 5;
    const int strip   = w & 3;
    const int colbase = (w >> 2) * 128;

    // Stage the whole z tile (contiguous 32 KiB, already bf16 -- pure copy).
#if __has_builtin(__builtin_amdgcn_global_load_async_to_lds_b128) && \
    __has_builtin(__builtin_amdgcn_s_wait_asynccnt)
    {
        // CDNA5 async copy: GLOBAL_LOAD_ASYNC_TO_LDS_B128, ASYNCcnt-tracked.
        // Builtin signature (from probe): (v4i AS1* src, v4i AS3* dst, imm, imm)
        typedef __attribute__((address_space(1))) v4i GV4;
        typedef __attribute__((address_space(3))) v4i LV4;
        GV4* g = (GV4*)(v4i*)(void*)const_cast<unsigned short*>(zb);
        LV4* l = (LV4*)(v4i*)&zs[0][0];
#pragma unroll
        for (int i = 0; i < 8; ++i) {
            int idx = i * 256 + t;  // 2048 x 16B = 32 KiB
            __builtin_amdgcn_global_load_async_to_lds_b128(g + idx, l + idx, 0, 0);
        }
        __builtin_amdgcn_s_wait_asynccnt(0);
    }
#else
    {
        const unsigned int* src = (const unsigned int*)zb;
        unsigned int* dst = (unsigned int*)&zs[0][0];
#pragma unroll 4
        for (int i = 0; i < 32; ++i) {
            dst[i * 256 + t] = src[i * 256 + t];
        }
    }
#endif

    v8f acc[8];
#pragma unroll
    for (int n = 0; n < 8; ++n) acc[n] = {};

    const int arow = strip * 16 + (lane & 15);
    const int hb   = lane >> 4;

    for (int kc = 0; kc < DD; kc += 32) {
        // Stage Wu chunk [32 x 256] transposed -> wt[c][k], 2x2 patches.
#pragma unroll
        for (int i = 0; i < 8; ++i) {
            int p  = i * 256 + t;
            int k2 = p >> 7;
            int d2 = p & 127;
            float2 r0 = *(const float2*)&wu[(kc + 2 * k2) * CC + c0 + 2 * d2];
            float2 r1 = *(const float2*)&wu[(kc + 2 * k2 + 1) * CC + c0 + 2 * d2];
            *(unsigned*)&wt[2 * d2][2 * k2]     = pack_bf16x2(r0.x, r1.x);
            *(unsigned*)&wt[2 * d2 + 1][2 * k2] = pack_bf16x2(r0.y, r1.y);
        }
        if (kc + 32 < DD) {
            __builtin_prefetch(&wu[(kc + 32) * CC + c0 + t], 0, 1);
        }
        __syncthreads();

        Frag af;
        af.q[0] = *(const uint4*)&zs[arow][kc + hb * 8];
        af.q[1] = *(const uint4*)&zs[arow][kc + hb * 8 + 16];

#pragma unroll
        for (int n = 0; n < 8; ++n) {
            int col = colbase + n * 16 + (lane & 15);
            int kh  = hb * 8;
            Frag bf;
            bf.q[0] = *(const uint4*)&wt[col][kh];
            bf.q[1] = *(const uint4*)&wt[col][kh + 16];
            acc[n] = __builtin_amdgcn_wmma_f32_16x16x32_bf16(
                false, af.v, false, bf.v, (short)0, acc[n], false, false);
        }
        __syncthreads();
    }

    // Epilogue: write f32 output
#pragma unroll
    for (int n = 0; n < 8; ++n) {
        int col = colbase + n * 16 + (lane & 15);
#pragma unroll
        for (int v = 0; v < 8; ++v) {
            int row = strip * 16 + v + 8 * hb;
            ub[row * CC + col] = acc[n][v];
        }
    }
}

extern "C" void kernel_launch(void* const* d_in, const int* in_sizes, int n_in,
                              void* d_out, int out_size, void* d_ws, size_t ws_size,
                              hipStream_t stream) {
    const float* x            = (const float*)d_in[0];
    const int*   expert_index = (const int*)d_in[1];
    const float* down_w       = (const float*)d_in[2];
    const float* down_b       = (const float*)d_in[3];
    const float* up_w         = (const float*)d_in[4];
    float* out = (float*)d_out;

    // z workspace: M*B*S*D bf16 = 16 MiB
    unsigned short* z_ws = (unsigned short*)d_ws;

    dim3 grid1(MM * BB, SS / 64);
    adapter_down_silu<<<grid1, 256, 0, stream>>>(x, expert_index, down_w, down_b, z_ws);

    dim3 grid2(MM * BB, SS / 64, CC / 256);
    adapter_up<<<grid2, 256, 0, stream>>>(z_ws, expert_index, up_w, out);
}
